// NMT_89816356094335
// MI455X (gfx1250) — matrix operand order repaired and checked by reading, not compiled, not run on hardware
//
#include <hip/hip_runtime.h>
#include <hip/hip_bf16.h>
#include <math.h>
#include <stdint.h>

// ---------------- model dims (from reference) ----------------
#define HDIM   512
#define EDIM   512
#define BB     32
#define SS     64
#define TT     48
#define VTV    32000
#define FOURH  2048
#define DWIN2  50.0f   // D*D/2, D=10

typedef __attribute__((ext_vector_type(16))) __bf16 v16bf;
typedef __attribute__((ext_vector_type(8)))  float  v8f;

// ---------------- WMMA helpers (CDNA5 bf16 16x16x32, f32 acc) ----------------
__device__ __forceinline__ v8f wmma_bf16(v16bf a, v16bf b, v8f c) {
  // 8 args: (neg_a, A, neg_b, B, c_mod, C, reuse_a, reuse_b)
  return __builtin_amdgcn_wmma_f32_16x16x32_bf16(false, a, false, b, (short)0, c, false, false);
}

// A fragment (16x32, bf16) from an f32 row-major [M,K] matrix, converting on load.
// Lane L: row = mbase + (L&15); halves 0..7 = K k0+off..k0+off+7, halves 8..15 = k0+16+off..,
// off = (L&16)?8:0.
__device__ __forceinline__ v16bf load_a_f32(const float* base, int ld, int mbase, int k0) {
  const int lane = threadIdx.x & 31;
  const float* p = base + (size_t)(mbase + (lane & 15)) * ld + k0 + ((lane & 16) ? 8 : 0);
  v16bf a;
#pragma unroll
  for (int i = 0; i < 8; ++i) a[i] = (__bf16)p[i];
#pragma unroll
  for (int i = 0; i < 8; ++i) a[8 + i] = (__bf16)p[16 + i];
  return a;
}

__device__ __forceinline__ v16bf load_a_bf16(const __bf16* base, int ld, int mbase, int k0) {
  const int lane = threadIdx.x & 31;
  const __bf16* p = base + (size_t)(mbase + (lane & 15)) * ld + k0 + ((lane & 16) ? 8 : 0);
  v16bf a;
#pragma unroll
  for (int i = 0; i < 8; ++i) a[i] = p[i];
#pragma unroll
  for (int i = 0; i < 8; ++i) a[8 + i] = p[16 + i];
  return a;
}

// B fragment (32x16, bf16) where B[k][n] = W[nbase+n][k0+k], W row-major with row stride ldw.
// Lane L: column n = L&15; halves 0..7 = K k0+off..k0+off+7, halves 8..15 = k0+16+off.., off = (L&16)?8:0.
__device__ __forceinline__ v16bf load_b_bf16(const __bf16* W, int ldw, int nbase, int k0) {
  const int lane = threadIdx.x & 31;
  const __bf16* p = W + (size_t)(nbase + (lane & 15)) * ldw + k0 + ((lane & 16) ? 8 : 0);
  v16bf b;
#pragma unroll
  for (int i = 0; i < 8; ++i) b[i] = p[i];
#pragma unroll
  for (int i = 0; i < 8; ++i) b[8 + i] = p[16 + i];
  return b;
}

// ---------------- elementwise utility kernels ----------------
__global__ void k_cvt_bf16(const float* __restrict__ x, __bf16* __restrict__ y, int n) {
  int i = blockIdx.x * 256 + threadIdx.x;
  if (i < n) y[i] = (__bf16)x[i];
}

__global__ void k_zero_f32(float* __restrict__ p, int n) {
  int i = blockIdx.x * 256 + threadIdx.x;
  if (i < n) p[i] = 0.0f;
}

// gather embedding rows -> bf16 [rows, 512]
__global__ void k_embed(const int* __restrict__ ids, const float* __restrict__ emb,
                        __bf16* __restrict__ out, int rows) {
  int idx = blockIdx.x * 256 + threadIdx.x;
  if (idx < rows * EDIM) {
    int r = idx >> 9, e = idx & 511;
    out[idx] = (__bf16)emb[(size_t)ids[r] * EDIM + e];
  }
}

// ---------------- generic WMMA GEMM: C[M,N] = A_bf16[M,K] @ W_bf16[N,:]^T + bias ----------------
// grid = (N/64, M/16), block = 32 (one wave, 4 N-tiles per wave).
__global__ __launch_bounds__(32) void k_gemm(const __bf16* __restrict__ A,
                                             const __bf16* __restrict__ W, int ldw,
                                             const float* __restrict__ bias,
                                             float* __restrict__ C, int ldc, int K) {
  const int nbase = blockIdx.x * 64;
  const int mbase = blockIdx.y * 16;
  const int lane = threadIdx.x;
  v8f acc[4];
#pragma unroll
  for (int t = 0; t < 4; ++t)
#pragma unroll
    for (int i = 0; i < 8; ++i) acc[t][i] = 0.0f;

  for (int k0 = 0; k0 < K; k0 += 32) {
    v16bf a = load_a_bf16(A, K, mbase, k0);
#pragma unroll
    for (int t = 0; t < 4; ++t) {
      v16bf b = load_b_bf16(W, ldw, nbase + t * 16, k0);
      acc[t] = wmma_bf16(a, b, acc[t]);
    }
  }
  const int rofs = (lane & 16) ? 8 : 0;
  const int cl = lane & 15;
#pragma unroll
  for (int t = 0; t < 4; ++t) {
    int col = nbase + t * 16 + cl;
    float bv = bias ? bias[col] : 0.0f;
#pragma unroll
    for (int i = 0; i < 8; ++i)
      C[(size_t)(mbase + rofs + i) * ldc + col] = acc[t][i] + bv;
  }
}

// ---------------- one LSTM timestep ----------------
// z[b, g*512+j] = Zpre + bias + x1 @ W1^T (+ x2 @ W2^T), gates i,f,g,o; update c, h.
// grid = 32 blocks (16 h-cols each), block = 128 (wave w handles gate w).
__global__ __launch_bounds__(128) void k_lstm(const float* __restrict__ Zpre,     // [32,2048] or null
                                              const float* __restrict__ bias,     // [2048] or null
                                              const float* __restrict__ x1,       // [32,512]
                                              const __bf16* __restrict__ W1, int ldw1,
                                              const float* __restrict__ x2,       // [32,512] or null
                                              const __bf16* __restrict__ W2, int ldw2,
                                              float* __restrict__ c_state,        // [32,512] in/out
                                              float* __restrict__ h_out,          // [32,512]
                                              float* __restrict__ hseq_f32,       // opt
                                              __bf16* __restrict__ hseq_bf16) {   // opt
  __shared__ float zs[4][BB][17];
  const int wv = threadIdx.x >> 5;     // gate index
  const int lane = threadIdx.x & 31;
  const int j0 = blockIdx.x * 16;
  const int nbase = wv * HDIM + j0;

  v8f acc0, acc1;
#pragma unroll
  for (int i = 0; i < 8; ++i) { acc0[i] = 0.0f; acc1[i] = 0.0f; }

  for (int k0 = 0; k0 < HDIM; k0 += 32) {
    v16bf b = load_b_bf16(W1, ldw1, nbase, k0);
    v16bf a0 = load_a_f32(x1, HDIM, 0, k0);
    v16bf a1 = load_a_f32(x1, HDIM, 16, k0);
    acc0 = wmma_bf16(a0, b, acc0);
    acc1 = wmma_bf16(a1, b, acc1);
  }
  if (x2) {
    for (int k0 = 0; k0 < HDIM; k0 += 32) {
      v16bf b = load_b_bf16(W2, ldw2, nbase, k0);
      v16bf a0 = load_a_f32(x2, HDIM, 0, k0);
      v16bf a1 = load_a_f32(x2, HDIM, 16, k0);
      acc0 = wmma_bf16(a0, b, acc0);
      acc1 = wmma_bf16(a1, b, acc1);
    }
  }

  const int rofs = (lane & 16) ? 8 : 0;
  const int cl = lane & 15;
  const int col = nbase + cl;
  const float bv = bias ? bias[col] : 0.0f;
#pragma unroll
  for (int i = 0; i < 8; ++i) {
    int b0 = rofs + i;
    zs[wv][b0][cl] = acc0[i] + bv + (Zpre ? Zpre[(size_t)b0 * FOURH + col] : 0.0f);
    int b1 = 16 + rofs + i;
    zs[wv][b1][cl] = acc1[i] + bv + (Zpre ? Zpre[(size_t)b1 * FOURH + col] : 0.0f);
  }
  __syncthreads();

  for (int idx = threadIdx.x; idx < BB * 16; idx += 128) {
    int b = idx >> 4, j = idx & 15;
    float zi = zs[0][b][j], zf = zs[1][b][j], zg = zs[2][b][j], zo = zs[3][b][j];
    float ig = 1.0f / (1.0f + expf(-zi));
    float fg = 1.0f / (1.0f + expf(-zf));
    float gg = tanhf(zg);
    float og = 1.0f / (1.0f + expf(-zo));
    int jg = j0 + j;
    size_t o = (size_t)b * HDIM + jg;
    float c2 = fg * c_state[o] + ig * gg;
    float h2 = og * tanhf(c2);
    c_state[o] = c2;
    h_out[o] = h2;
    if (hseq_f32)  hseq_f32[o] = h2;
    if (hseq_bf16) hseq_bf16[o] = (__bf16)h2;
  }
}

// ---------------- Luong local attention (per decoder step) ----------------
__global__ __launch_bounds__(256) void k_attn(const float* __restrict__ yt_buf,     // h1 [32,512]
                                              const float* __restrict__ ench,       // [S,B,H]
                                              const int* __restrict__ encode_len,
                                              const float* __restrict__ W_ht2tan,   // [512,512]
                                              const float* __restrict__ W_tan2pt,   // [512]
                                              const float* __restrict__ W_ct2ht,    // [512,1024]
                                              float* __restrict__ ht_out,           // [32,512]
                                              __bf16* __restrict__ decout_bf16) {   // [32,512] slice
  __shared__ float yt[HDIM], tv[HDIM], sc[SS], at[SS], ct[HDIM], red[256];
  __shared__ float pt_s, mx_s, sum_s;
  const int b = blockIdx.x;
  const int tid = threadIdx.x;

  for (int i = tid; i < HDIM; i += 256) yt[i] = yt_buf[(size_t)b * HDIM + i];
  __syncthreads();

  for (int j = tid; j < HDIM; j += 256) {
    const float* w = W_ht2tan + (size_t)j * HDIM;
    float s = 0.0f;
    for (int k = 0; k < HDIM; ++k) s += yt[k] * w[k];
    tv[j] = tanhf(s);
  }
  __syncthreads();

  float part = 0.0f;
  for (int k = tid; k < HDIM; k += 256) part += tv[k] * W_tan2pt[k];
  red[tid] = part;
  __syncthreads();
  for (int s = 128; s > 0; s >>= 1) { if (tid < s) red[tid] += red[tid + s]; __syncthreads(); }
  const int elen = encode_len[b];
  if (tid == 0) pt_s = (1.0f / (1.0f + expf(-red[0]))) * (float)elen;

  if (tid < SS) {
    const float* eh = ench + ((size_t)tid * BB + b) * HDIM;
    float s = 0.0f;
    for (int k = 0; k < HDIM; ++k) s += yt[k] * eh[k];
    sc[tid] = (tid < elen) ? s : -3.0e38f;
  }
  __syncthreads();
  if (tid == 0) {
    float m = -3.0e38f;
    for (int s = 0; s < SS; ++s) m = fmaxf(m, sc[s]);
    mx_s = m;
  }
  __syncthreads();
  if (tid < SS) at[tid] = expf(sc[tid] - mx_s);
  __syncthreads();
  if (tid == 0) {
    float s = 0.0f;
    for (int i = 0; i < SS; ++i) s += at[i];
    sum_s = s;
  }
  __syncthreads();
  if (tid < SS) {
    float d = (float)tid - pt_s;
    at[tid] = (at[tid] / sum_s) * expf(-(d * d) / DWIN2);
  }
  __syncthreads();

  for (int h = tid; h < HDIM; h += 256) {
    float s = 0.0f;
    for (int ss = 0; ss < SS; ++ss) s += at[ss] * ench[((size_t)ss * BB + b) * HDIM + h];
    ct[h] = s;
  }
  __syncthreads();

  for (int j = tid; j < HDIM; j += 256) {
    const float* w = W_ct2ht + (size_t)j * (2 * HDIM);
    float s = 0.0f;
    for (int k = 0; k < HDIM; ++k) s += ct[k] * w[k];
    for (int k = 0; k < HDIM; ++k) s += yt[k] * w[HDIM + k];
    float h2 = tanhf(s);
    ht_out[(size_t)b * HDIM + j] = h2;
    decout_bf16[(size_t)b * HDIM + j] = (__bf16)h2;
  }
}

// ---------------- vocab projection + streaming logsumexp + target gather ----------------
// grid = 94 blocks (16 rows), block = 256 (8 waves split 32000 vocab in 64-col chunks).
// A tile (16x512 bf16 = 16KB) staged LDS via CDNA5 async loads, then register-resident.
#define AROW 520   // padded row stride in halves (1040B): conflict-free b128 reads
__global__ __launch_bounds__(256) void k_vocab(const __bf16* __restrict__ Abf,   // [1536,512]
                                               const __bf16* __restrict__ Wf,    // [32000,512]
                                               const int* __restrict__ target,   // [T,B]
                                               float* __restrict__ lp) {         // [1504]
  __shared__ __bf16 Alds[16 * AROW];
  __shared__ float sm[8][16], ss_[8][16], st[8][16];
  const int mbase = blockIdx.x * 16;
  const int wv = threadIdx.x >> 5;
  const int lane = threadIdx.x & 31;
  const int rofs = (lane & 16) ? 8 : 0;
  const int cl = lane & 15;

  // ---- async-stage A tile into LDS: 1024 x 16B transfers, 4 per thread ----
  {
    const uint32_t lds_base = (uint32_t)(uintptr_t)(&Alds[0]);
#pragma unroll
    for (int i = 0; i < 4; ++i) {
      int c = threadIdx.x + 256 * i;      // 16B chunk id, 0..1023
      int row = c >> 6, inrow = c & 63;   // 64 chunks per 512-half row
      uint32_t ldsa = lds_base + (uint32_t)(row * AROW + inrow * 8) * 2u;
      uint64_t ga = (uint64_t)(uintptr_t)(Abf + (size_t)(mbase + row) * HDIM + inrow * 8);
      asm volatile("global_load_async_to_lds_b128 %0, %1, off"
                   :: "v"(ldsa), "v"(ga) : "memory");
    }
    asm volatile("s_wait_asynccnt 0x0" ::: "memory");
    __syncthreads();
  }

  // ---- pull A fragments from LDS into registers (held for all 500 chunks) ----
  v16bf afrag[16];
  {
    const __bf16* arow = &Alds[(lane & 15) * AROW];
    const int off = (lane & 16) ? 8 : 0;
#pragma unroll
    for (int kk = 0; kk < 16; ++kk) {
      const __bf16* p = arow + kk * 32 + off;
#pragma unroll
      for (int i = 0; i < 8; ++i) afrag[kk][i] = p[i];
#pragma unroll
      for (int i = 0; i < 8; ++i) afrag[kk][8 + i] = p[16 + i];
    }
  }

  int tok[8];
#pragma unroll
  for (int i = 0; i < 8; ++i) tok[i] = target[mbase + rofs + i + BB];  // target[t+1, b]

  float m[8], s[8], tg[8];
#pragma unroll
  for (int i = 0; i < 8; ++i) { m[i] = -3.0e38f; s[i] = 0.0f; tg[i] = 0.0f; }

  const int NCH = VTV / 64;  // 500 chunks of 64 columns
  for (int c = wv; c < NCH; c += 8) {
    const int n0 = c * 64;
    if (c + 8 < NCH) __builtin_prefetch(Wf + (size_t)((c + 8) * 64 + cl) * HDIM, 0, 1);
    v8f acc[4];
#pragma unroll
    for (int t = 0; t < 4; ++t)
#pragma unroll
      for (int i = 0; i < 8; ++i) acc[t][i] = 0.0f;

#pragma unroll
    for (int kk = 0; kk < 16; ++kk) {
      const int k0 = kk * 32;
#pragma unroll
      for (int t = 0; t < 4; ++t) {
        v16bf b = load_b_bf16(Wf, HDIM, n0 + t * 16, k0);
        acc[t] = wmma_bf16(afrag[kk], b, acc[t]);
      }
    }

    // online logsumexp: lane-local max over 4 tiles first (8 shfl + 5 exp per row)
#pragma unroll
    for (int i = 0; i < 8; ++i) {
      float v0 = acc[0][i], v1 = acc[1][i], v2 = acc[2][i], v3 = acc[3][i];
#pragma unroll
      for (int t = 0; t < 4; ++t)
        if (n0 + t * 16 + cl == tok[i]) tg[i] = acc[t][i];
      float vm = fmaxf(fmaxf(v0, v1), fmaxf(v2, v3));
#pragma unroll
      for (int d = 1; d < 16; d <<= 1) vm = fmaxf(vm, __shfl_xor(vm, d, 32));
      float nm = fmaxf(m[i], vm);
      float e = expf(v0 - nm) + expf(v1 - nm) + expf(v2 - nm) + expf(v3 - nm);
#pragma unroll
      for (int d = 1; d < 16; d <<= 1) e += __shfl_xor(e, d, 32);
      s[i] = s[i] * expf(m[i] - nm) + e;
      m[i] = nm;
    }
  }
  // reduce captured target logits across the 16-lane half
#pragma unroll
  for (int i = 0; i < 8; ++i) {
    float t = tg[i];
#pragma unroll
    for (int d = 1; d < 16; d <<= 1) t += __shfl_xor(t, d, 32);
    tg[i] = t;
  }

  if (lane == 0)
#pragma unroll
    for (int i = 0; i < 8; ++i) { sm[wv][i] = m[i]; ss_[wv][i] = s[i]; st[wv][i] = tg[i]; }
  if (lane == 16)
#pragma unroll
    for (int i = 0; i < 8; ++i) { sm[wv][8 + i] = m[i]; ss_[wv][8 + i] = s[i]; st[wv][8 + i] = tg[i]; }
  __syncthreads();

  if (threadIdx.x < 16) {
    int r = threadIdx.x;
    float M = -3.0e38f;
    for (int w = 0; w < 8; ++w) M = fmaxf(M, sm[w][r]);
    float S = 0.0f, Tg = 0.0f;
    for (int w = 0; w < 8; ++w) { S += ss_[w][r] * expf(sm[w][r] - M); Tg += st[w][r]; }
    lp[mbase + r] = Tg - (M + logf(S));
  }
}

__global__ void k_final(const float* __restrict__ lp, float* __restrict__ out) {
  int b = threadIdx.x;
  if (b < BB) {
    float s = 0.0f;
    for (int t = 0; t < TT - 1; ++t) s += lp[t * BB + b];
    out[b] = s;
  }
}

// ---------------- host orchestration ----------------
extern "C" void kernel_launch(void* const* d_in, const int* in_sizes, int n_in,
                              void* d_out, int out_size, void* d_ws, size_t ws_size,
                              hipStream_t stream) {
  (void)in_sizes; (void)n_in; (void)out_size; (void)ws_size;
  const int*   source     = (const int*)d_in[0];
  const int*   target     = (const int*)d_in[1];
  const int*   encode_len = (const int*)d_in[2];
  const float* src_emb    = (const float*)d_in[3];
  const float* tar_emb    = (const float*)d_in[4];
  const float* enc_Wih0   = (const float*)d_in[5];
  const float* enc_Whh0   = (const float*)d_in[6];
  const float* enc_b0     = (const float*)d_in[7];
  const float* enc_Wih1   = (const float*)d_in[8];
  const float* enc_Whh1   = (const float*)d_in[9];
  const float* enc_b1     = (const float*)d_in[10];
  const float* dec_Wih0   = (const float*)d_in[11];
  const float* dec_Whh0   = (const float*)d_in[12];
  const float* dec_b0     = (const float*)d_in[13];
  const float* dec_Wih1   = (const float*)d_in[14];
  const float* dec_Whh1   = (const float*)d_in[15];
  const float* dec_b1     = (const float*)d_in[16];
  const float* W_ht2tan   = (const float*)d_in[17];
  const float* W_tan2pt   = (const float*)d_in[18];
  const float* W_ct2ht    = (const float*)d_in[19];
  const float* W_final    = (const float*)d_in[20];
  float* out = (float*)d_out;

  char* w = (char*)d_ws;
  auto abf = [&](size_t n) { __bf16* p = (__bf16*)w; w += ((n * 2 + 255) & ~(size_t)255); return p; };
  auto af  = [&](size_t n) { float*  p = (float*)w;  w += ((n * 4 + 255) & ~(size_t)255); return p; };

  __bf16* wbf_eWih0 = abf(FOURH * EDIM);
  __bf16* wbf_eWhh0 = abf(FOURH * HDIM);
  __bf16* wbf_eWih1 = abf(FOURH * HDIM);
  __bf16* wbf_eWhh1 = abf(FOURH * HDIM);
  __bf16* wbf_dWih0 = abf((size_t)FOURH * (EDIM + HDIM));
  __bf16* wbf_dWhh0 = abf(FOURH * HDIM);
  __bf16* wbf_dWih1 = abf(FOURH * HDIM);
  __bf16* wbf_dWhh1 = abf(FOURH * HDIM);
  __bf16* wbf_Wfin  = abf((size_t)VTV * HDIM);
  __bf16* Xs        = abf((size_t)SS * BB * EDIM);
  __bf16* Yt        = abf((size_t)TT * BB * EDIM);
  __bf16* H0seq     = abf((size_t)SS * BB * HDIM);
  __bf16* DecOut    = abf((size_t)TT * BB * HDIM);
  float*  Z         = af((size_t)SS * BB * FOURH);   // reused: enc pre0/pre1, dec Ypre
  float*  ench      = af((size_t)SS * BB * HDIM);
  float*  states    = af((size_t)7 * BB * HDIM);     // h0a,h0b,h1a,h1b,c0,c1,ht
  float*  lp        = af((size_t)(TT - 1) * BB);

  float* h0a = states + 0 * BB * HDIM;
  float* h0b = states + 1 * BB * HDIM;
  float* h1a = states + 2 * BB * HDIM;
  float* h1b = states + 3 * BB * HDIM;
  float* c0  = states + 4 * BB * HDIM;
  float* c1  = states + 5 * BB * HDIM;
  float* ht  = states + 6 * BB * HDIM;

  auto cvt = [&](const float* s, __bf16* d, size_t n) {
    k_cvt_bf16<<<dim3((unsigned)((n + 255) / 256)), 256, 0, stream>>>(s, d, (int)n);
  };
  cvt(enc_Wih0, wbf_eWih0, (size_t)FOURH * EDIM);
  cvt(enc_Whh0, wbf_eWhh0, (size_t)FOURH * HDIM);
  cvt(enc_Wih1, wbf_eWih1, (size_t)FOURH * HDIM);
  cvt(enc_Whh1, wbf_eWhh1, (size_t)FOURH * HDIM);
  cvt(dec_Wih0, wbf_dWih0, (size_t)FOURH * (EDIM + HDIM));
  cvt(dec_Whh0, wbf_dWhh0, (size_t)FOURH * HDIM);
  cvt(dec_Wih1, wbf_dWih1, (size_t)FOURH * HDIM);
  cvt(dec_Whh1, wbf_dWhh1, (size_t)FOURH * HDIM);
  cvt(W_final,  wbf_Wfin,  (size_t)VTV * HDIM);

  k_embed<<<dim3((SS * BB * EDIM + 255) / 256), 256, 0, stream>>>(source, src_emb, Xs, SS * BB);
  k_embed<<<dim3((TT * BB * EDIM + 255) / 256), 256, 0, stream>>>(target, tar_emb, Yt, TT * BB);

  int nstate = 7 * BB * HDIM;
  k_zero_f32<<<dim3((nstate + 255) / 256), 256, 0, stream>>>(states, nstate);

  // ===== encoder layer 0 =====
  k_gemm<<<dim3(FOURH / 64, (SS * BB) / 16), 32, 0, stream>>>(Xs, wbf_eWih0, EDIM, enc_b0, Z, FOURH, EDIM);
  float* hc = h0a; float* hn = h0b;
  for (int sstep = 0; sstep < SS; ++sstep) {
    k_lstm<<<32, 128, 0, stream>>>(Z + (size_t)sstep * BB * FOURH, nullptr,
                                   hc, wbf_eWhh0, HDIM, nullptr, nullptr, 0,
                                   c0, hn, nullptr, H0seq + (size_t)sstep * BB * HDIM);
    float* t = hc; hc = hn; hn = t;
  }
  // ===== encoder layer 1 =====
  k_gemm<<<dim3(FOURH / 64, (SS * BB) / 16), 32, 0, stream>>>(H0seq, wbf_eWih1, HDIM, enc_b1, Z, FOURH, HDIM);
  float* hc1 = h1a; float* hn1 = h1b;
  for (int sstep = 0; sstep < SS; ++sstep) {
    k_lstm<<<32, 128, 0, stream>>>(Z + (size_t)sstep * BB * FOURH, nullptr,
                                   hc1, wbf_eWhh1, HDIM, nullptr, nullptr, 0,
                                   c1, hn1, ench + (size_t)sstep * BB * HDIM, nullptr);
    float* t = hc1; hc1 = hn1; hn1 = t;
  }

  // ===== decoder =====
  k_gemm<<<dim3(FOURH / 64, (TT * BB) / 16), 32, 0, stream>>>(Yt, wbf_dWih0, EDIM + HDIM, dec_b0, Z, FOURH, EDIM);
  for (int t = 0; t < TT; ++t) {
    k_lstm<<<32, 128, 0, stream>>>(Z + (size_t)t * BB * FOURH, nullptr,
                                   hc, wbf_dWhh0, HDIM,
                                   ht, wbf_dWih0 + EDIM, EDIM + HDIM,
                                   c0, hn, nullptr, nullptr);
    { float* tm = hc; hc = hn; hn = tm; }
    k_lstm<<<32, 128, 0, stream>>>(nullptr, dec_b1,
                                   hc, wbf_dWih1, HDIM,
                                   hc1, wbf_dWhh1, HDIM,
                                   c1, hn1, nullptr, nullptr);
    { float* tm = hc1; hc1 = hn1; hn1 = tm; }
    k_attn<<<BB, 256, 0, stream>>>(hc1, ench, encode_len, W_ht2tan, W_tan2pt, W_ct2ht,
                                   ht, DecOut + (size_t)t * BB * HDIM);
  }

  // ===== vocab projection + logsumexp + gather + sum =====
  k_vocab<<<dim3(((TT - 1) * BB) / 16), 256, 0, stream>>>(DecOut, wbf_Wfin, target, lp);
  k_final<<<1, 32, 0, stream>>>(lp, out);
}
